// PairwiseSimilarity_50560355008576
// MI455X (gfx1250) — matrix operand order repaired and checked by reading, not compiled, not run on hardware
//
#include <hip/hip_runtime.h>

// ---------------------------------------------------------------------------
// Pairwise RBF similarity: out[i,j] = exp(-||z_i - z_j||^2), z: [8192,128] f32
//   d2 = sq[i] + sq[j] - 2 * (z @ z^T)[i,j]
//
// MI455X roofline:
//   * Output 256 MB f32 @ 23.3 TB/s -> ~11 us floor; stored non-temporal.
//   * Gram via v_wmma_f32_16x16x32_f16, split precision z = hi + lo (f16):
//     g ~= hi*hi + hi*lo + lo*hi -> ~22-bit input mantissa at f16 WMMA rate.
//     Split is precomputed once into d_ws (no conversion VALU in hot loop).
//   * LDS-staged 128x128 block tiles: 128 KB L2 read per 64 KB output (2:1),
//     keeping L2 read demand (~47 TB/s) below plausible L2 bandwidth so the
//     NT output stream stays the bottleneck.
//   * Register budget: __launch_bounds__(256,2) + k-loop unroll 2 keeps the
//     wave <= 256 VGPRs -> no s_set_vgpr_msb toggling, and 2 blocks/WGP
//     (4 waves/SIMD) matching the 2x136 KB LDS plan.
// ---------------------------------------------------------------------------

typedef _Float16 half_t;
typedef __attribute__((ext_vector_type(16))) _Float16 v16h;
typedef __attribute__((ext_vector_type(8)))  _Float16 v8h;
typedef __attribute__((ext_vector_type(4)))  _Float16 v4h;
typedef __attribute__((ext_vector_type(8)))  float    v8f;

#define N_TOK 8192
#define D_DIM 128
#define LDS_STRIDE 136   // 128 halves + 8-half (16B) pad: stride = 4 banks mod 64

// ---- Kernel 0: split z into f16 hi + f16 residual, plus row norms ---------
__global__ __launch_bounds__(256) void pack_kernel(const float* __restrict__ z,
                                                   half_t* __restrict__ zh,
                                                   half_t* __restrict__ zl,
                                                   float* __restrict__ sq) {
    const int row = blockIdx.x * 256 + threadIdx.x;
    const float4* zr = reinterpret_cast<const float4*>(z + (size_t)row * D_DIM);
    v4h* hr = reinterpret_cast<v4h*>(zh + (size_t)row * D_DIM);
    v4h* lr = reinterpret_cast<v4h*>(zl + (size_t)row * D_DIM);
    float s = 0.f;
#pragma unroll
    for (int t = 0; t < D_DIM / 4; ++t) {
        float4 v = zr[t];
        s = fmaf(v.x, v.x, s);
        s = fmaf(v.y, v.y, s);
        s = fmaf(v.z, v.z, s);
        s = fmaf(v.w, v.w, s);
        v4h h, l;
        h[0] = (half_t)v.x; l[0] = (half_t)(v.x - (float)h[0]);
        h[1] = (half_t)v.y; l[1] = (half_t)(v.y - (float)h[1]);
        h[2] = (half_t)v.z; l[2] = (half_t)(v.z - (float)h[2]);
        h[3] = (half_t)v.w; l[3] = (half_t)(v.w - (float)h[3]);
        hr[t] = h;
        lr[t] = l;
    }
    sq[row] = s;
}

// ---- LDS fragment loaders (ISA 16-bit WMMA VGPR layouts, wave32) ----------
// A (16x32, MxK): lane m = lane&15 is row M; hi = lane>>4 selects K runs
//   [k0+hi*8 .. +7] and [k0+16+hi*8 .. +7]  -> two ds_load_b128.
__device__ __forceinline__ v16h lds_frag_a(const half_t* base, int k0, int hi) {
    union { v16h v; struct { v8h a, b; } s; } u;
    u.s.a = *reinterpret_cast<const v8h*>(base + k0 + hi * 8);
    u.s.b = *reinterpret_cast<const v8h*>(base + k0 + 16 + hi * 8);
    return u.v;
}
// B (32x16, KxN): lane m = lane&15 is col N; K runs [k0+hi*8 .. +7]
//   and [k0+16+hi*8 .. +7] (two ds_load_b128), matching A.
__device__ __forceinline__ v16h lds_frag_b(const half_t* base, int k0, int hi) {
    union { v16h v; struct { v8h a, b; } s; } u;
    const half_t* p = base + k0 + hi * 8;
    u.s.a = *reinterpret_cast<const v8h*>(p);
    u.s.b = *reinterpret_cast<const v8h*>(p + 16);
    return u.v;
}

#define WMMA_F16(A, B, C) \
    __builtin_amdgcn_wmma_f32_16x16x32_f16(false, (A), false, (B), (short)0, (C), false, false)

// ---- Kernel 1: 8 waves/block; wave -> 32x64 strip; block -> 128x128 -------
__global__ __launch_bounds__(256, 2) void gram_rbf_kernel(const half_t* __restrict__ zh,
                                                          const half_t* __restrict__ zl,
                                                          const float* __restrict__ sq,
                                                          float* __restrict__ out) {
    __shared__ half_t sAh[128 * LDS_STRIDE];
    __shared__ half_t sAl[128 * LDS_STRIDE];
    __shared__ half_t sBh[128 * LDS_STRIDE];
    __shared__ half_t sBl[128 * LDS_STRIDE];

    const int tid  = threadIdx.x;
    const int lane = tid & 31;
    const int wave = tid >> 5;
    const int m    = lane & 15;
    const int hi   = lane >> 4;

    const int rowPanel = blockIdx.y * 128;
    const int colPanel = blockIdx.x * 128;

    // --- Stage A/B panels (hi+lo) into LDS: coalesced b128 copies ----------
    // Each panel: 128 rows x 256B = 2048 16B-chunks; 8 chunks per thread.
#pragma unroll
    for (int i = 0; i < 8; ++i) {
        const int id  = tid + i * 256;      // 0..2047
        const int row = id >> 4;            // panel-local row
        const int kc  = (id & 15) * 16;     // byte offset within row
        const size_t gA = (size_t)(rowPanel + row) * D_DIM;
        const size_t gB = (size_t)(colPanel + row) * D_DIM;
        const int    so = row * (LDS_STRIDE * 2) + kc;

        *reinterpret_cast<uint4*>((char*)sAh + so) =
            *reinterpret_cast<const uint4*>((const char*)(zh + gA) + kc);
        *reinterpret_cast<uint4*>((char*)sAl + so) =
            *reinterpret_cast<const uint4*>((const char*)(zl + gA) + kc);
        *reinterpret_cast<uint4*>((char*)sBh + so) =
            *reinterpret_cast<const uint4*>((const char*)(zh + gB) + kc);
        *reinterpret_cast<uint4*>((char*)sBl + so) =
            *reinterpret_cast<const uint4*>((const char*)(zl + gB) + kc);
    }
    __syncthreads();

    // --- Wave tile: 32 rows x 64 cols --------------------------------------
    const int ar0 = ((wave >> 1) * 32 + m)      * LDS_STRIDE;  // A frag rows
    const int ar1 = ((wave >> 1) * 32 + 16 + m) * LDS_STRIDE;

    v8f c[2][4];
#pragma unroll
    for (int i = 0; i < 2; ++i)
#pragma unroll
        for (int j = 0; j < 4; ++j) c[i][j] = (v8f){};

    // unroll 2 (not 4): keeps the load-hoisting window small enough that the
    // wave stays under 256 VGPRs (no s_set_vgpr_msb, 2 blocks/WGP resident).
#pragma unroll 2
    for (int k0 = 0; k0 < D_DIM; k0 += 32) {
        const v16h ah0 = lds_frag_a(sAh + ar0, k0, hi);
        const v16h ah1 = lds_frag_a(sAh + ar1, k0, hi);
        const v16h al0 = lds_frag_a(sAl + ar0, k0, hi);
        const v16h al1 = lds_frag_a(sAl + ar1, k0, hi);

#pragma unroll
        for (int j = 0; j < 4; ++j) {
            const int br = ((wave & 1) * 64 + j * 16 + m) * LDS_STRIDE;
            const v16h bh = lds_frag_b(sBh + br, k0, hi);
            const v16h bl = lds_frag_b(sBl + br, k0, hi);

            // g += hi*hi + hi*lo + lo*hi (lo*lo ~ 2^-22 of signal, dropped)
            c[0][j] = WMMA_F16(ah0, bh, c[0][j]);
            c[0][j] = WMMA_F16(ah0, bl, c[0][j]);
            c[0][j] = WMMA_F16(al0, bh, c[0][j]);
            c[1][j] = WMMA_F16(ah1, bh, c[1][j]);
            c[1][j] = WMMA_F16(ah1, bl, c[1][j]);
            c[1][j] = WMMA_F16(al1, bh, c[1][j]);
        }
    }

    // --- Epilogue: d2 = sr + sc - 2g, clamp, exp, NT store -----------------
    // C/D layout: VGPR r -> row (tile_row + hi*8 + r), col (tile_col + m).
    const int rowBase = rowPanel + (wave >> 1) * 32;
    const int col0    = colPanel + (wave & 1) * 64 + m;
    float scj[4];
#pragma unroll
    for (int j = 0; j < 4; ++j) scj[j] = sq[col0 + j * 16];

#pragma unroll
    for (int i = 0; i < 2; ++i) {
#pragma unroll
        for (int r = 0; r < 8; ++r) {
            const int row = rowBase + i * 16 + hi * 8 + r;
            const float sr = sq[row];
            float* __restrict__ orow = out + (size_t)row * N_TOK;
#pragma unroll
            for (int j = 0; j < 4; ++j) {
                float g  = c[i][j][r];
                float d2 = fmaxf(sr + scj[j] - 2.0f * g, 0.0f);
                // SIGMA == 1.0
                __builtin_nontemporal_store(__expf(-d2), orow + col0 + j * 16);
            }
        }
    }
}

extern "C" void kernel_launch(void* const* d_in, const int* in_sizes, int n_in,
                              void* d_out, int out_size, void* d_ws, size_t ws_size,
                              hipStream_t stream) {
    const float* z   = (const float*)d_in[0];
    float*       out = (float*)d_out;

    // Workspace layout: sq (32 KB) | z_hi (2 MB) | z_lo (2 MB)
    float*  sq = (float*)d_ws;
    half_t* zh = (half_t*)((char*)d_ws + 32 * 1024);
    half_t* zl = zh + (size_t)N_TOK * D_DIM;

    pack_kernel<<<N_TOK / 256, 256, 0, stream>>>(z, zh, zl, sq);

    dim3 grid(N_TOK / 128, N_TOK / 128);   // (64, 64) blocks, 8 waves each
    gram_rbf_kernel<<<grid, 256, 0, stream>>>(zh, zl, sq, out);
}
